// FlashMultiHeadAttention_45414984187905
// MI455X (gfx1250) — matrix operand order repaired and checked by reading, not compiled, not run on hardware
//
#include <hip/hip_runtime.h>
#include <hip/hip_bf16.h>

// CDNA5 (gfx1250) wave32 WMMA flash-attention MHA.
//   D = A(16x32 bf16) x B(32x16 bf16) + C(16x16 f32), v_wmma_f32_16x16x32_bf16

typedef __bf16 v16bf __attribute__((ext_vector_type(16)));
typedef float  v8f   __attribute__((ext_vector_type(8)));

#define HID   1024
#define HEADS 16
#define DH    64
#define BATCH 4
#define SEQ   2048

__device__ __forceinline__ __bf16 f2bf(float f) {
  union { float f; unsigned u; } v; v.f = f;
  unsigned r = v.u + 0x7FFFu + ((v.u >> 16) & 1u);   // round-to-nearest-even
  unsigned short h = (unsigned short)(r >> 16);
  union { unsigned short u; __bf16 b; } o; o.u = h; return o.b;
}

// ---------------------------------------------------------------------------
// One-pass f32 -> bf16 streaming conversion (weights). n must be /4.
// ---------------------------------------------------------------------------
__global__ __launch_bounds__(256)
void cvt_f32_bf16(const float* __restrict__ x, __bf16* __restrict__ y, int n) {
  const int i = (blockIdx.x * 256 + threadIdx.x) * 4;
  if (i + 3 < n) {
    const float4 f = *(const float4*)(x + i);
    y[i + 0] = f2bf(f.x);
    y[i + 1] = f2bf(f.y);
    y[i + 2] = f2bf(f.z);
    y[i + 3] = f2bf(f.w);
  }
}

// ---------------------------------------------------------------------------
// Per-wave 16(M) x 128(N) GEMM tile: Y = X @ W^T + bias, W pre-converted bf16.
// MODE 0: Y bf16 row-major [M,HID]        (Q / K projections)
// MODE 1: Y bf16 transposed [B,H,Dh,S]    (V projection -> Vt)
// MODE 2: Y f32  row-major [M,HID]        (output projection -> d_out)
// XBF16: X is bf16 (attention output) vs f32 (raw activations)
// ---------------------------------------------------------------------------
template <int MODE, bool XBF16>
__global__ __launch_bounds__(32)
void gemm_wmma(const void* __restrict__ Xv, const __bf16* __restrict__ Wb,
               const float* __restrict__ bias, void* __restrict__ Yv) {
  const int lane = threadIdx.x;
  const int half = lane >> 4;       // 0: lanes 0-15, 1: lanes 16-31
  const int lrow = lane & 15;
  const int m0 = blockIdx.x * 16;
  const int n0 = blockIdx.y * 128;
  const int row = m0 + lrow;        // A: lane (mod 16) = M row

  v8f acc[8];
#pragma unroll
  for (int j = 0; j < 8; ++j) acc[j] = v8f{};

  for (int k0 = 0; k0 < HID; k0 += 32) {
    // ---- A tile 16x32 bf16: chunks K = k0+(half?8:0) and +16
    v16bf a;
    const int c0 = k0 + (half ? 8 : 0);
    const int c1 = c0 + 16;
    if (XBF16) {
      const __bf16* X = (const __bf16*)Xv;
      *(uint4*)&a       = *(const uint4*)(X + (size_t)row * HID + c0);
      *((uint4*)&a + 1) = *(const uint4*)(X + (size_t)row * HID + c1);
    } else {
      const float* X = (const float*)Xv;
      const float* p0 = X + (size_t)row * HID + c0;
      const float* p1 = X + (size_t)row * HID + c1;
      __bf16* d = (__bf16*)&a;
#pragma unroll
      for (int i = 0; i < 8; ++i) d[i]     = f2bf(p0[i]);
#pragma unroll
      for (int i = 0; i < 8; ++i) d[8 + i] = f2bf(p1[i]);
    }
#pragma unroll
    for (int j = 0; j < 8; ++j) {
      // ---- B tile 32x16: lane = N column = W row; chunks K = k0+(half?8:0) and +16
      const int n = n0 + j * 16 + lrow;
      const __bf16* wp = Wb + (size_t)n * HID + k0 + (half ? 8 : 0);
      v16bf bb;
      *(uint4*)&bb       = *(const uint4*)(wp);
      *((uint4*)&bb + 1) = *(const uint4*)(wp + 16);
      acc[j] = __builtin_amdgcn_wmma_f32_16x16x32_bf16(
          false, a, false, bb, (short)0, acc[j], false, false);
    }
  }

  // ---- epilogue: C layout -> element v of acc j is (m0+v+8*half, n0+j*16+lrow)
#pragma unroll
  for (int j = 0; j < 8; ++j) {
#pragma unroll
    for (int v = 0; v < 8; ++v) {
      const int m = m0 + v + 8 * half;
      const int n = n0 + j * 16 + lrow;
      const float val = acc[j][v] + bias[n];
      if (MODE == 0) {
        ((__bf16*)Yv)[(size_t)m * HID + n] = f2bf(val);
      } else if (MODE == 1) {
        const int bb = m >> 11, s = m & (SEQ - 1);
        const int h = n >> 6, dd = n & (DH - 1);
        ((__bf16*)Yv)[(((size_t)bb * HEADS + h) * DH + dd) * SEQ + s] = f2bf(val);
      } else {
        ((float*)Yv)[(size_t)m * HID + n] = val;
      }
    }
  }
}

// ---------------------------------------------------------------------------
// Flash attention: one wave handles one (b, h, 16-row q tile); online softmax
// over 32-wide k blocks up to the causal diagonal.
// ---------------------------------------------------------------------------
__global__ __launch_bounds__(32)
void flash_attn(const __bf16* __restrict__ Qp, const __bf16* __restrict__ Kp,
                const __bf16* __restrict__ Vt, const float* __restrict__ rt,
                __bf16* __restrict__ O) {
  __shared__ __align__(16) __bf16 lds_p[16 * 32];   // P tile, row-major 16(q) x 32(k)

  const int lane = threadIdx.x;
  const int half = lane >> 4;
  const int lrow = lane & 15;
  const int q0 = blockIdx.x * 16;
  const int h  = blockIdx.y;
  const int b  = blockIdx.z;

  // Q in A layout, split over d: qa0 (d 0..31), qa1 (d 32..63)
  const __bf16* qrow = Qp + ((size_t)b * SEQ + q0 + lrow) * HID + h * DH;
  v16bf qa0, qa1;
  {
    const int c0 = (half ? 8 : 0), c1 = 16 + (half ? 8 : 0);
    *(uint4*)&qa0       = *(const uint4*)(qrow + c0);
    *((uint4*)&qa0 + 1) = *(const uint4*)(qrow + c1);
    *(uint4*)&qa1       = *(const uint4*)(qrow + 32 + c0);
    *((uint4*)&qa1 + 1) = *(const uint4*)(qrow + 32 + c1);
  }

  v8f o0{}, o1{}, o2{}, o3{};
  float m_run[8], l_run[8];
  int rowoff[8];
#pragma unroll
  for (int v = 0; v < 8; ++v) {
    m_run[v] = -__builtin_inff();
    l_run[v] = 0.f;
    rowoff[v] = (q0 + v + 8 * half) * SEQ;           // rt row base (fits 32-bit)
  }

  const float scale = 0.125f;                         // 1/sqrt(64)
  const float* rtb  = rt + (size_t)b * SEQ * SEQ;     // relative_time[b,0,:,:]
  const __bf16* vbase = Vt + ((size_t)b * HEADS + h) * DH * SEQ;
  const int kend = q0 + 16;                           // causal horizon (exclusive)

  for (int k0 = 0; k0 < kend; k0 += 32) {
    // ---- scores: two 16-column tiles, each K(=d)-depth 64 via 2 WMMAs
    v8f sc[2];
#pragma unroll
    for (int t = 0; t < 2; ++t) {
      const int kc = k0 + t * 16 + lrow;              // column owned by lane
      const __bf16* krow = Kp + ((size_t)b * SEQ + kc) * HID + h * DH + (half ? 8 : 0);
      v16bf kb0, kb1;
      *(uint4*)&kb0       = *(const uint4*)(krow);
      *((uint4*)&kb0 + 1) = *(const uint4*)(krow + 16);
      *(uint4*)&kb1       = *(const uint4*)(krow + 32);
      *((uint4*)&kb1 + 1) = *(const uint4*)(krow + 48);
      v8f s{};
      s = __builtin_amdgcn_wmma_f32_16x16x32_bf16(false, qa0, false, kb0, (short)0, s, false, false);
      s = __builtin_amdgcn_wmma_f32_16x16x32_bf16(false, qa1, false, kb1, (short)0, s, false, false);
      sc[t] = s;
    }

    // ---- bias (pre-scale, per reference), scale, causal mask; per-row max
    float comb[8];
#pragma unroll
    for (int v = 0; v < 8; ++v) {
      const int row = q0 + v + 8 * half;
#pragma unroll
      for (int t = 0; t < 2; ++t) {
        const int col = k0 + t * 16 + lrow;
        const int cl  = col <= row ? col : row;       // keep bias read in-bounds
        float s = (sc[t][v] + rtb[rowoff[v] + cl]) * scale;
        if (col > row) s = -__builtin_inff();
        sc[t][v] = s;
      }
      comb[v] = fmaxf(sc[0][v], sc[1][v]);
    }
#pragma unroll
    for (int off = 1; off < 16; off <<= 1) {
#pragma unroll
      for (int v = 0; v < 8; ++v)
        comb[v] = fmaxf(comb[v], __shfl_xor(comb[v], off, 32));
    }

    // ---- online softmax update
#pragma unroll
    for (int v = 0; v < 8; ++v) {
      const float nm    = fmaxf(m_run[v], comb[v]);
      const float alpha = __expf(m_run[v] - nm);
      m_run[v] = nm;
      float rs = 0.f;
#pragma unroll
      for (int t = 0; t < 2; ++t) {
        const float p = __expf(sc[t][v] - nm);        // masked -inf -> 0
        sc[t][v] = p;
        rs += p;
      }
#pragma unroll
      for (int off = 1; off < 16; off <<= 1) rs += __shfl_xor(rs, off, 32);
      l_run[v] = l_run[v] * alpha + rs;
      o0[v] *= alpha; o1[v] *= alpha; o2[v] *= alpha; o3[v] *= alpha;
    }

    // ---- transpose P through LDS: C layout -> A layout bf16
    __syncthreads();
#pragma unroll
    for (int v = 0; v < 8; ++v) {
      const int r = v + 8 * half;
      lds_p[r * 32 + lrow]      = f2bf(sc[0][v]);
      lds_p[r * 32 + 16 + lrow] = f2bf(sc[1][v]);
    }
    __syncthreads();
    v16bf pa;
    {
      const int c0 = (half ? 8 : 0), c1 = 16 + (half ? 8 : 0);
      *(uint4*)&pa       = *(const uint4*)(&lds_p[lrow * 32 + c0]);
      *((uint4*)&pa + 1) = *(const uint4*)(&lds_p[lrow * 32 + c1]);
    }
    __syncthreads();

    // ---- O += P x V over 4 d-tiles; Vt gives contiguous 32B B-tile loads
#pragma unroll
    for (int j = 0; j < 4; ++j) {
      const __bf16* vp = vbase + (size_t)(j * 16 + lrow) * SEQ + k0 + (half ? 8 : 0);
      v16bf vb;
      *(uint4*)&vb       = *(const uint4*)(vp);
      *((uint4*)&vb + 1) = *(const uint4*)(vp + 16);
      v8f acc = (j == 0) ? o0 : (j == 1) ? o1 : (j == 2) ? o2 : o3;
      acc = __builtin_amdgcn_wmma_f32_16x16x32_bf16(false, pa, false, vb, (short)0, acc, false, false);
      if (j == 0) o0 = acc; else if (j == 1) o1 = acc; else if (j == 2) o2 = acc; else o3 = acc;
    }
  }

  // ---- normalize and emit attention output (bf16 row-major [B,S,HID])
#pragma unroll
  for (int v = 0; v < 8; ++v) {
    const float inv = 1.0f / l_run[v];
    const int m = q0 + v + 8 * half;
    __bf16* orow = O + ((size_t)b * SEQ + m) * HID + h * DH;
    orow[ 0 + lrow] = f2bf(o0[v] * inv);
    orow[16 + lrow] = f2bf(o1[v] * inv);
    orow[32 + lrow] = f2bf(o2[v] * inv);
    orow[48 + lrow] = f2bf(o3[v] * inv);
  }
}

// ---------------------------------------------------------------------------
extern "C" void kernel_launch(void* const* d_in, const int* in_sizes, int n_in,
                              void* d_out, int out_size, void* d_ws, size_t ws_size,
                              hipStream_t stream) {
  const float* query = (const float*)d_in[0];
  const float* key   = (const float*)d_in[1];
  const float* value = (const float*)d_in[2];
  const float* rt    = (const float*)d_in[3];
  /* d_in[4] = attn_mask (causal tril) — recomputed from indices in-kernel */
  const float* Wq = (const float*)d_in[5];
  const float* bq = (const float*)d_in[6];
  const float* Wk = (const float*)d_in[7];
  const float* bk = (const float*)d_in[8];
  const float* Wv = (const float*)d_in[9];
  const float* bv = (const float*)d_in[10];
  const float* Wo = (const float*)d_in[11];
  const float* bo = (const float*)d_in[12];
  (void)in_sizes; (void)n_in; (void)out_size; (void)ws_size;

  const size_t M  = (size_t)BATCH * SEQ;              // 8192 tokens
  const size_t WN = (size_t)HID * HID;                // 1M weight elements
  __bf16* Qp  = (__bf16*)d_ws;                         // [B,S,HID]
  __bf16* Kp  = Qp + M * HID;                          // [B,S,HID]
  __bf16* Vt  = Kp + M * HID;                          // [B,H,Dh,S]
  __bf16* Oa  = Vt + M * HID;                          // [B,S,HID]
  __bf16* Wqb = Oa + M * HID;                          // 4x [HID,HID] bf16
  __bf16* Wkb = Wqb + WN;
  __bf16* Wvb = Wkb + WN;
  __bf16* Wob = Wvb + WN;

  // 1) pre-convert weights once (streaming, ~17 MB total)
  {
    dim3 cblk(256), cgrid((unsigned)(WN / (256 * 4)));
    cvt_f32_bf16<<<cgrid, cblk, 0, stream>>>(Wq, Wqb, (int)WN);
    cvt_f32_bf16<<<cgrid, cblk, 0, stream>>>(Wk, Wkb, (int)WN);
    cvt_f32_bf16<<<cgrid, cblk, 0, stream>>>(Wv, Wvb, (int)WN);
    cvt_f32_bf16<<<cgrid, cblk, 0, stream>>>(Wo, Wob, (int)WN);
  }

  // 2) projections
  dim3 blk(32);
  dim3 gproj((unsigned)(M / 16), HID / 128, 1);
  gemm_wmma<0, false><<<gproj, blk, 0, stream>>>(query, Wqb, bq, Qp);
  gemm_wmma<0, false><<<gproj, blk, 0, stream>>>(key,   Wkb, bk, Kp);
  gemm_wmma<1, false><<<gproj, blk, 0, stream>>>(value, Wvb, bv, Vt);

  // 3) flash attention
  dim3 gattn(SEQ / 16, HEADS, BATCH);
  flash_attn<<<gattn, blk, 0, stream>>>(Qp, Kp, Vt, rt, Oa);

  // 4) output projection
  gemm_wmma<2, true><<<gproj, blk, 0, stream>>>(Oa, Wob, bo, (float*)d_out);
}
